// TopologicalMemory_12017318494889
// MI455X (gfx1250) — matrix operand order repaired and checked by reading, not compiled, not run on hardware
//
#include <hip/hip_runtime.h>
#include <math.h>

// Problem constants (match the reference)
#define MEMN   8192
#define FEATN  512
#define BATCH  16
#define TAU_NEW 0.85f
#define D_MIN   1.5f
#define ALPHAF  0.5f
#define EPSV    1e-8f

typedef __attribute__((ext_vector_type(16))) _Float16 v16h;
typedef __attribute__((ext_vector_type(8)))  float    v8f;

// ---------------------------------------------------------------------------
// Kernel 1: row norms of the (original) feature matrix.  fn[m] = ||feat[m]||
// 8 rows per 256-thread block, one wave32 per row.
// ---------------------------------------------------------------------------
__global__ __launch_bounds__(256) void row_norms_k(const float* __restrict__ feat,
                                                   float* __restrict__ fn) {
    const int lane = threadIdx.x & 31;
    const int wave = threadIdx.x >> 5;
    const int row  = blockIdx.x * 8 + wave;
    const float* r = feat + (size_t)row * FEATN;
    float s = 0.f;
    for (int k = lane; k < FEATN; k += 32) { float v = r[k]; s += v * v; }
    for (int off = 16; off; off >>= 1) s += __shfl_xor(s, off, 32);
    if (lane == 0) fn[row] = sqrtf(s);
}

// ---------------------------------------------------------------------------
// Kernel 2: batched similarity GEMM via WMMA.
//   simsRaw[n][m] = dot(feat[m], h[n])   (un-normalized)
// One wave per 16-row tile of feat; K=512 consumed in 16 WMMA f16 k=32 steps.
// A fragment (16x32 f16): lanes 0-15 rows M=lane, halves 0..7 -> K=k0..k0+7,
//   halves 8..15 -> K=k0+16..k0+23; lanes 16-31 same rows, K bases +8.
// B fragment (32x16 f16): lane n=lane&15, same K layout as A:
//   halves 0..7 -> K=k0..k0+7, halves 8..15 -> K=k0+16..k0+23; lanes 16-31 K bases +8.
// C/D (16x16 f32): lanes 0-15 -> M=j (vgpr j), lanes 16-31 -> M=8+j; N=lane&15.
// ---------------------------------------------------------------------------
__global__ __launch_bounds__(32) void sims_gemm_k(const float* __restrict__ feat,  // [M][FEAT]
                                                  const float* __restrict__ h,     // [B][FEAT]
                                                  float* __restrict__ simsRaw) {   // [B][M]
    const int tile = blockIdx.x;       // 0..511
    const int lane = threadIdx.x;      // 0..31
    const int hi   = lane & 15;
    const int sel  = lane >> 4;        // 0 or 1 (half-wave)

    const float* arow = feat + (size_t)(tile * 16 + hi) * FEATN;  // A row = feat row
    const float* brow = h    + (size_t)hi * FEATN;                // B col n = h row n

    v8f acc = {};
    for (int k0 = 0; k0 < FEATN; k0 += 32) {
        v16h a, b;
        const int a0 = k0 + sel * 8;        // halves 0..7
        const int a1 = k0 + 16 + sel * 8;   // halves 8..15
#pragma unroll
        for (int e = 0; e < 8; ++e) {
            a[e]     = (_Float16)arow[a0 + e];
            a[e + 8] = (_Float16)arow[a1 + e];
        }
#pragma unroll
        for (int e = 0; e < 8; ++e) {
            b[e]     = (_Float16)brow[a0 + e];
            b[e + 8] = (_Float16)brow[a1 + e];
        }

        acc = __builtin_amdgcn_wmma_f32_16x16x32_f16(
            /*neg_a=*/false, a, /*neg_b=*/false, b,
            /*c_mod=*/(short)0, acc, /*reuse_a=*/false, /*reuse_b=*/false);
    }
    // scatter the 16x16 tile: simsRaw[n][tile*16 + m]
    const int n     = hi;
    const int mrow0 = tile * 16 + sel * 8;
    float* dst = simsRaw + (size_t)n * MEMN + mrow0;
#pragma unroll
    for (int j = 0; j < 8; ++j) dst[j] = acc[j];
}

// ---------------------------------------------------------------------------
// Kernel 3: the sequential 16-step scan, one 512-thread block (16 waves).
// State deltas live in LDS: <=16 modified feature rows (+norms, positions) and
// a deduped list of <=32 adjacency cells set to 1.0.  Global feat/pos/adj are
// read-only (inputs must not be mutated).
// ---------------------------------------------------------------------------
__global__ __launch_bounds__(512) void topo_scan_k(
    const float* __restrict__ h,              // [B][FEAT]
    const float* __restrict__ cpos,           // [B][3]
    const unsigned char* __restrict__ dones,  // [B] (bool8)
    const float* __restrict__ feat,           // [M][FEAT]
    const float* __restrict__ npos,           // [M][3]
    const float* __restrict__ adj,            // [M][M]
    const int* __restrict__ pNum,
    const int* __restrict__ pPtr,
    const int* __restrict__ lastIn,           // [B]
    const float* __restrict__ simsRaw,        // [B][M]
    const float* __restrict__ fn,             // [M]
    float* __restrict__ out)                  // [B]
{
    const int tid  = threadIdx.x;
    const int lane = tid & 31;
    const int wave = tid >> 5;

    __shared__ float modFeat[16][FEATN];   // 32 KB: updated feature rows
    __shared__ float modPos[16][3];
    __shared__ float modNorm[16];
    __shared__ int   modRow[16];
    __shared__ float hn[BATCH];
    __shared__ float red[512];
    __shared__ int   redi[512];
    __shared__ int   edgeR[40], edgeC[40];
    __shared__ int   s_modCount, s_edgeCount, s_n, s_p;
    __shared__ int   s_lastv[BATCH];
    __shared__ float s_modSims[16];
    __shared__ int   s_widx, s_slot, s_doWrite, s_ema, s_done, s_cur;

    if (tid == 0) {
        s_modCount = 0; s_edgeCount = 0;
        s_n = pNum[0];  s_p = pPtr[0];
    }
    if (tid < BATCH) s_lastv[tid] = lastIn[tid];
    {   // hn[w] = ||h[w]||  (wave w handles row w; exactly 16 waves)
        const float* r = h + (size_t)wave * FEATN;
        float s = 0.f;
        for (int k = lane; k < FEATN; k += 32) { float v = r[k]; s += v * v; }
        for (int off = 16; off; off >>= 1) s += __shfl_xor(s, off, 32);
        if (lane == 0) hn[wave] = sqrtf(s);
    }
    __syncthreads();

    for (int i = 0; i < BATCH; ++i) {
        const int  nCur = s_n, pCur = s_p, mc = s_modCount;
        const float hden = fmaxf(hn[i], EPSV);

        // --- Phase 1: masked argmax over unmodified rows (precomputed sims)
        float bv = -2.0f; int bi = 0;
        for (int m = tid; m < MEMN; m += 512) {
            if (m >= nCur) continue;
            bool mod = false;
            for (int s = 0; s < mc; ++s) if (modRow[s] == m) { mod = true; break; }
            if (mod) continue;
            float v = simsRaw[(size_t)i * MEMN + m] / (fmaxf(fn[m], EPSV) * hden);
            if (v > bv || (v == bv && m < bi)) { bv = v; bi = m; }
        }
        red[tid] = bv; redi[tid] = bi;
        __syncthreads();
        for (int s = 256; s > 0; s >>= 1) {
            if (tid < s) {
                float ov = red[tid + s]; int oi = redi[tid + s];
                if (ov > red[tid] || (ov == red[tid] && oi < redi[tid])) {
                    red[tid] = ov; redi[tid] = oi;
                }
            }
            __syncthreads();
        }
        // --- Phase 2: fresh sims for the <=16 modified rows (wave s -> slot s)
        if (wave < mc) {
            const float* mf = modFeat[wave];
            const float* hr = h + (size_t)i * FEATN;
            float s = 0.f;
            for (int k = lane; k < FEATN; k += 32) s += mf[k] * hr[k];
            for (int off = 16; off; off >>= 1) s += __shfl_xor(s, off, 32);
            if (lane == 0)
                s_modSims[wave] = s / (fmaxf(modNorm[wave], EPSV) * hden);
        }
        __syncthreads();

        // --- Phase 3: scalar decision (thread 0)
        if (tid == 0) {
            float mv = red[0]; int mi = redi[0];
            for (int s = 0; s < mc; ++s) {
                int r = modRow[s];
                if (r >= nCur) continue;
                float v = s_modSims[s];
                if (v > mv || (v == mv && r < mi)) { mv = v; mi = r; }
            }
            const int best = mi; const float max_sim = mv;

            int bslot = -1;
            for (int s = 0; s < mc; ++s) if (modRow[s] == best) { bslot = s; break; }
            float bp0 = (bslot >= 0) ? modPos[bslot][0] : npos[best * 3 + 0];
            float bp1 = (bslot >= 0) ? modPos[bslot][1] : npos[best * 3 + 1];
            float bp2 = (bslot >= 0) ? modPos[bslot][2] : npos[best * 3 + 2];
            float dx = cpos[i * 3 + 0] - bp0, dy = cpos[i * 3 + 1] - bp1,
                  dz = cpos[i * 3 + 2] - bp2;
            float dist = sqrtf(dx * dx + dy * dy + dz * dz);

            const int done       = dones[i] ? 1 : 0;
            const int empty      = (nCur < 1);
            const int should_add = (max_sim < TAU_NEW) || (dist > D_MIN);
            const int has_room   = (nCur < MEMN);
            const int widx = empty ? 0 : (should_add ? (has_room ? nCur : pCur) : best);
            const int ema  = (!empty) && (!should_add);
            s_widx = widx; s_ema = ema; s_done = done; s_cur = widx;

            int doWrite = !done, slot = -1;       // done => byte-identical no-op write
            if (doWrite) {
                for (int s = 0; s < mc; ++s) if (modRow[s] == widx) { slot = s; break; }
                if (slot < 0) { slot = s_modCount; s_modCount = slot + 1; modRow[slot] = widx; }
            }
            s_slot = slot; s_doWrite = doWrite;

            if (!done) {
                if (empty || (should_add && has_room)) s_n = nCur + 1;
                if ((!empty) && should_add && (!has_room)) s_p = (pCur + 1) % MEMN;
            }
            // adjacency edge list (deduped; value always set to 1.0)
            const int last = s_lastv[i];
            const int cur  = widx;
            if ((!done) && (last != -1) && (last != cur)) {
                int lc = last < 0 ? 0 : (last > MEMN - 1 ? MEMN - 1 : last);
                int pr[2] = {lc, cur}, pc[2] = {cur, lc};
                for (int e = 0; e < 2; ++e) {
                    bool found = false;
                    for (int q = 0; q < s_edgeCount; ++q)
                        if (edgeR[q] == pr[e] && edgeC[q] == pc[e]) { found = true; break; }
                    if (!found && s_edgeCount < 40) {
                        edgeR[s_edgeCount] = pr[e];
                        edgeC[s_edgeCount] = pc[e];
                        s_edgeCount++;
                    }
                }
            }
            s_lastv[i] = done ? -1 : cur;
        }
        __syncthreads();

        // --- Phase 4: apply the single-row feature/position write (into LDS)
        if (s_doWrite) {   // block-uniform branch
            const int slot = s_slot, widx = s_widx, ema = s_ema;
            const bool isNew = (slot >= mc);
            const float* hr = h + (size_t)i * FEATN;
            for (int k = tid; k < FEATN; k += 512) {
                float oldv = isNew ? feat[(size_t)widx * FEATN + k] : modFeat[slot][k];
                modFeat[slot][k] = ema ? (ALPHAF * oldv + (1.f - ALPHAF) * hr[k]) : hr[k];
            }
            if (tid < 3) {
                float oldp = isNew ? npos[widx * 3 + tid] : modPos[slot][tid];
                modPos[slot][tid] =
                    ema ? (ALPHAF * oldp + (1.f - ALPHAF) * cpos[i * 3 + tid])
                        : cpos[i * 3 + tid];
            }
            __syncthreads();
            float s = 0.f;
            for (int k = tid; k < FEATN; k += 512) { float v = modFeat[slot][k]; s += v * v; }
            red[tid] = s;
            __syncthreads();
            for (int st = 256; st > 0; st >>= 1) {
                if (tid < st) red[tid] += red[tid + st];
                __syncthreads();
            }
            if (tid == 0) modNorm[slot] = sqrtf(red[0]);
        }
        __syncthreads();

        // --- Phase 5: topo = sum(adj_new[cur]) = sum(adj_orig[cur]) + corrections
        {
            const int cur = s_cur;
            const float* row = adj + (size_t)cur * MEMN;
            float s = 0.f;
            for (int m = tid; m < MEMN; m += 512) s += row[m];
            red[tid] = s;
            __syncthreads();
            for (int st = 256; st > 0; st >>= 1) {
                if (tid < st) red[tid] += red[tid + st];
                __syncthreads();
            }
            if (tid == 0) {
                float topo = red[0];
                for (int q = 0; q < s_edgeCount; ++q)
                    if (edgeR[q] == cur)
                        topo += 1.0f - adj[(size_t)cur * MEMN + edgeC[q]];
                out[i] = s_done ? 0.f : topo;
            }
        }
        __syncthreads();
    }
}

// ---------------------------------------------------------------------------
extern "C" void kernel_launch(void* const* d_in, const int* in_sizes, int n_in,
                              void* d_out, int out_size, void* d_ws, size_t ws_size,
                              hipStream_t stream) {
    (void)in_sizes; (void)n_in; (void)out_size; (void)ws_size;
    const float*         h     = (const float*)d_in[0];         // [16][512]
    const float*         cpos  = (const float*)d_in[1];         // [16][3]
    const unsigned char* dones = (const unsigned char*)d_in[2]; // [16] bool8
    const float*         feat  = (const float*)d_in[3];         // [8192][512]
    const float*         npos  = (const float*)d_in[4];         // [8192][3]
    const float*         adj   = (const float*)d_in[5];         // [8192][8192]
    const int*           pNum  = (const int*)d_in[6];           // scalar
    const int*           pPtr  = (const int*)d_in[7];           // scalar
    const int*           lastv = (const int*)d_in[8];           // [16]

    // Workspace: simsRaw (16*8192 f32 = 512 KB) + fn (8192 f32 = 32 KB)
    float* simsRaw = (float*)d_ws;
    float* fn      = simsRaw + (size_t)BATCH * MEMN;

    row_norms_k<<<MEMN / 8, 256, 0, stream>>>(feat, fn);
    sims_gemm_k<<<MEMN / 16, 32, 0, stream>>>(feat, h, simsRaw);
    topo_scan_k<<<1, 512, 0, stream>>>(h, cpos, dones, feat, npos, adj,
                                       pNum, pPtr, lastv, simsRaw, fn,
                                       (float*)d_out);
}
